// SelfCausalAttention_44289702756786
// MI455X (gfx1250) — matrix operand order repaired and checked by reading, not compiled, not run on hardware
//
#include <hip/hip_runtime.h>
#include <hip/hip_bf16.h>

#define N_HEAD   12
#define HEAD_DIM 64
#define N_EMBED  768
#define SEQ_T    2048
#define BATCH    2

typedef _Float16 f16;
typedef _Float16 v8h  __attribute__((ext_vector_type(8)));
typedef _Float16 v16h __attribute__((ext_vector_type(16)));
typedef float    v8f  __attribute__((ext_vector_type(8)));

static __device__ __forceinline__ v8f wmma16x16x32(v16h a, v16h b, v8f c) {
  // D = A(16x32 f16) x B(32x16 f16) + C(16x16 f32)
  return __builtin_amdgcn_wmma_f32_16x16x32_f16(false, a, false, b, (short)0, c, false, false);
}

static __device__ __forceinline__ v16h joinv(v8h lo, v8h hi) {
  return __builtin_shufflevector(lo, hi, 0,1,2,3,4,5,6,7,8,9,10,11,12,13,14,15);
}

// A fragment (16x32, f16). rowptr = base of this lane's row (row-major, K-contiguous).
// Layout (matches load_b_frag): lanes0-15 hold K=[k+0..15];
// lanes16-31 hold K=[k+16..31].
static __device__ __forceinline__ v16h load_a_frag(const f16* rowptr, int k, int lane) {
  int hs = (lane >> 4) * 16;
  v8h lo = *(const v8h*)(rowptr + k + hs);
  v8h hi = *(const v8h*)(rowptr + k + hs + 8);
  return joinv(lo, hi);
}

// B fragment (32x16, f16). colptr = base of this lane's column data, K-contiguous
// (operand stored transposed). lanes0-15: K=[k+0..15], lanes16-31: K=[k+16..31].
static __device__ __forceinline__ v16h load_b_frag(const f16* colptr, int k, int lane) {
  int ko = (lane >> 4) * 16;
  v8h lo = *(const v8h*)(colptr + k + ko);
  v8h hi = *(const v8h*)(colptr + k + ko + 8);
  return joinv(lo, hi);
}

// ---------------- conversion kernels ----------------

__global__ void f32_to_f16_kernel(const float* __restrict__ in, f16* __restrict__ out, int n) {
  int i = blockIdx.x * blockDim.x + threadIdx.x;
  if (i < n) out[i] = (f16)in[i];
}

// in: f32 [K][N] row-major  ->  out: f16 [N][K] (transposed)
__global__ void transpose_f32_to_f16_kernel(const float* __restrict__ in, f16* __restrict__ out,
                                            int K, int N) {
  int i = blockIdx.x * blockDim.x + threadIdx.x;
  if (i >= K * N) return;
  int k = i % K;
  int n = i / K;
  out[(size_t)n * K + k] = (f16)in[(size_t)k * N + n];
}

// ---------------- WMMA GEMM: C[M,N] = A[M,K] @ Bt[N,K]^T + bias ----------------
// One wave per 16(M) x 64(N) output tile: the A fragment is reused across four
// WMMAs per k-step (4 WMMAs : 10 b128 loads), so the loop is matrix-issue bound
// rather than load-issue bound. 4 waves per block -> 256-wide N panel per block.

__global__ void gemm_wmma_kernel(const f16* __restrict__ A, const f16* __restrict__ Bt,
                                 const float* __restrict__ bias, float* __restrict__ C,
                                 int M, int N, int K) {
  int lane = threadIdx.x & 31;
  int wave = threadIdx.x >> 5;
  int l15 = lane & 15;
  int lh  = lane >> 4;
  int tn = (blockIdx.x * 4 + wave) * 64;
  int tm = blockIdx.y * 16;

  const f16* arow = A  + (size_t)(tm + l15) * K;
  const f16* bc0  = Bt + (size_t)(tn + 0  + l15) * K;
  const f16* bc1  = Bt + (size_t)(tn + 16 + l15) * K;
  const f16* bc2  = Bt + (size_t)(tn + 32 + l15) * K;
  const f16* bc3  = Bt + (size_t)(tn + 48 + l15) * K;

  v8f acc0 = {}, acc1 = {}, acc2 = {}, acc3 = {};
  for (int k = 0; k < K; k += 32) {
    // prefetch one k-panel ahead (speculative: dropped safely past end)
    __builtin_prefetch(arow + k + 256, 0, 1);
    __builtin_prefetch(bc0 + k + 256, 0, 1);
    __builtin_prefetch(bc2 + k + 256, 0, 1);
    v16h a  = load_a_frag(arow, k, lane);
    v16h b0 = load_b_frag(bc0, k, lane);
    v16h b1 = load_b_frag(bc1, k, lane);
    v16h b2 = load_b_frag(bc2, k, lane);
    v16h b3 = load_b_frag(bc3, k, lane);
    acc0 = wmma16x16x32(a, b0, acc0);
    acc1 = wmma16x16x32(a, b1, acc1);
    acc2 = wmma16x16x32(a, b2, acc2);
    acc3 = wmma16x16x32(a, b3, acc3);
  }

  #pragma unroll
  for (int v = 0; v < 8; ++v) {
    int row = tm + v + lh * 8;            // C layout: vgpr v, lanes16-31 -> M = v+8
    float* crow = C + (size_t)row * N + tn + l15;
    crow[0]  = acc0[v] + bias[tn + l15];
    crow[16] = acc1[v] + bias[tn + 16 + l15];
    crow[32] = acc2[v] + bias[tn + 32 + l15];
    crow[48] = acc3[v] + bias[tn + 48 + l15];
  }
}

// ---------------- RoPE + head split ----------------
// qkv f32 [B,T,3E] -> Qh,Kh f16 [B,H,T,64] (roped), Vt f16 [B,H,64,T] (transposed)

__global__ void rope_split_kernel(const float* __restrict__ qkv,
                                  f16* __restrict__ Qh, f16* __restrict__ Kh,
                                  f16* __restrict__ Vt) {
  int i = blockIdx.x * blockDim.x + threadIdx.x;   // B*T*H*32 threads
  int j = i & 31;                                  // freq pair index
  int h = (i >> 5) % N_HEAD;
  int t = (i / (32 * N_HEAD)) % SEQ_T;
  int b = i / (32 * N_HEAD * SEQ_T);
  if (b >= BATCH) return;

  size_t base = ((size_t)(b * SEQ_T + t)) * (3 * N_EMBED) + h * HEAD_DIM + 2 * j;
  float qe = qkv[base],               qo = qkv[base + 1];
  float ke = qkv[base + N_EMBED],     ko = qkv[base + N_EMBED + 1];
  float ve = qkv[base + 2 * N_EMBED], vo = qkv[base + 2 * N_EMBED + 1];

  float inv_freq = __powf(10000.0f, -(float)(2 * j) / (float)HEAD_DIM);
  float ang = (float)t * inv_freq;
  float sn, cs;
  __sincosf(ang, &sn, &cs);

  size_t hb  = (size_t)(b * N_HEAD + h);
  size_t qko = (hb * SEQ_T + t) * HEAD_DIM + 2 * j;
  Qh[qko]     = (f16)(qe * cs - qo * sn);
  Qh[qko + 1] = (f16)(qe * sn + qo * cs);
  Kh[qko]     = (f16)(ke * cs - ko * sn);
  Kh[qko + 1] = (f16)(ke * sn + ko * cs);

  size_t vb = (hb * HEAD_DIM + 2 * j) * SEQ_T + t;
  Vt[vb]         = (f16)ve;
  Vt[vb + SEQ_T] = (f16)vo;
}

// ---------------- flash attention (one wave per 16-query tile) ----------------

__global__ void flash_attn_kernel(const f16* __restrict__ Qh, const f16* __restrict__ Kh,
                                  const f16* __restrict__ Vt, f16* __restrict__ Oh) {
  __shared__ __attribute__((aligned(16))) f16 pbuf[4][16 * 32];

  int lane = threadIdx.x & 31;
  int wave = threadIdx.x >> 5;
  int l15 = lane & 15;
  int lh  = lane >> 4;

  int qt = blockIdx.x * 4 + wave;
  int t0 = qt * 16;
  int bh = blockIdx.y;
  int h = bh % N_HEAD;
  int b = bh / N_HEAD;

  const f16* Q  = Qh + (size_t)bh * SEQ_T * HEAD_DIM;
  const f16* Kp = Kh + (size_t)bh * SEQ_T * HEAD_DIM;
  const f16* Vp = Vt + (size_t)bh * HEAD_DIM * SEQ_T;
  f16* myP = &pbuf[wave][0];

  const f16* qrow = Q + (size_t)(t0 + l15) * HEAD_DIM;
  v16h aQ0 = load_a_frag(qrow, 0, lane);    // d = 0..31
  v16h aQ1 = load_a_frag(qrow, 32, lane);   // d = 32..63

  float m[8], l[8];
  v8f o0 = {}, o1 = {}, o2 = {}, o3 = {};
  #pragma unroll
  for (int v = 0; v < 8; ++v) { m[v] = -__builtin_inff(); l[v] = 0.0f; }

  int nblk = (t0 + 15) / 32 + 1;            // causal: keys s <= t0+15
  for (int blk = 0; blk < nblk; ++blk) {
    int sb = blk * 32;

    // S = Q @ K^T for two 16-col score tiles (cols sb..sb+15, sb+16..sb+31)
    const f16* kr0 = Kp + (size_t)(sb + l15) * HEAD_DIM;
    const f16* kr1 = Kp + (size_t)(sb + 16 + l15) * HEAD_DIM;
    v8f sA = {}, sB = {};
    sA = wmma16x16x32(aQ0, load_b_frag(kr0, 0, lane), sA);
    sA = wmma16x16x32(aQ1, load_b_frag(kr0, 32, lane), sA);
    sB = wmma16x16x32(aQ0, load_b_frag(kr1, 0, lane), sB);
    sB = wmma16x16x32(aQ1, load_b_frag(kr1, 32, lane), sB);

    if (sb + 31 > t0) {                     // diagonal block: causal mask
      #pragma unroll
      for (int v = 0; v < 8; ++v) {
        int trow = t0 + v + lh * 8;
        if (sb + l15 > trow)      sA[v] = -__builtin_inff();
        if (sb + 16 + l15 > trow) sB[v] = -__builtin_inff();
      }
    }

    // online softmax update (row = v + 8*lh, 16 lanes per row per half-wave)
    #pragma unroll
    for (int v = 0; v < 8; ++v) {
      float rmax = fmaxf(sA[v], sB[v]);
      #pragma unroll
      for (int d = 1; d < 16; d <<= 1) rmax = fmaxf(rmax, __shfl_xor(rmax, d, 32));
      float mnew  = fmaxf(m[v], rmax);
      float alpha = __expf(m[v] - mnew);
      float pA = __expf(sA[v] - mnew);
      float pB = __expf(sB[v] - mnew);
      float rsum = pA + pB;
      #pragma unroll
      for (int d = 1; d < 16; d <<= 1) rsum += __shfl_xor(rsum, d, 32);
      l[v] = l[v] * alpha + rsum;
      m[v] = mnew;
      o0[v] *= alpha; o1[v] *= alpha; o2[v] *= alpha; o3[v] *= alpha;
      // stage P tile (16x32 row-major f16) in LDS for A-fragment reload
      myP[(v + lh * 8) * 32 + l15]      = (f16)pA;
      myP[(v + lh * 8) * 32 + 16 + l15] = (f16)pB;
    }
    asm volatile("s_wait_dscnt 0" ::: "memory");   // per-wave LDS store->load turnaround

    // O += P(16x32) @ V(32x64); Vt is [64][T] so B-operand is s-contiguous per column d
    v16h aP = load_a_frag(myP + l15 * 32, 0, lane);
    o0 = wmma16x16x32(aP, load_b_frag(Vp + (size_t)(0  + l15) * SEQ_T + sb, 0, lane), o0);
    o1 = wmma16x16x32(aP, load_b_frag(Vp + (size_t)(16 + l15) * SEQ_T + sb, 0, lane), o1);
    o2 = wmma16x16x32(aP, load_b_frag(Vp + (size_t)(32 + l15) * SEQ_T + sb, 0, lane), o2);
    o3 = wmma16x16x32(aP, load_b_frag(Vp + (size_t)(48 + l15) * SEQ_T + sb, 0, lane), o3);
  }

  // epilogue: O * (N_EMBED^-0.5 / l), write into merged [B,T,E] f16 layout for proj GEMM
  const float scale = 0.03608439182435161f;        // 768^-0.5
  #pragma unroll
  for (int v = 0; v < 8; ++v) {
    float inv = scale / l[v];
    int t = t0 + v + lh * 8;
    size_t ob = ((size_t)(b * SEQ_T + t)) * N_EMBED + h * HEAD_DIM + l15;
    Oh[ob]      = (f16)(o0[v] * inv);
    Oh[ob + 16] = (f16)(o1[v] * inv);
    Oh[ob + 32] = (f16)(o2[v] * inv);
    Oh[ob + 48] = (f16)(o3[v] * inv);
  }
}

// ---------------- launch ----------------

extern "C" void kernel_launch(void* const* d_in, const int* in_sizes, int n_in,
                              void* d_out, int out_size, void* d_ws, size_t ws_size,
                              hipStream_t stream) {
  const float* x      = (const float*)d_in[0];
  const float* w_attn = (const float*)d_in[1];
  const float* b_attn = (const float*)d_in[2];
  const float* w_proj = (const float*)d_in[3];
  const float* b_proj = (const float*)d_in[4];
  float* out = (float*)d_out;

  const int BT = BATCH * SEQ_T;  // 4096
  char* ws = (char*)d_ws;
  f16*   xh   = (f16*)ws;   ws += (size_t)BT * N_EMBED * 2;                 // x in f16
  f16*   waT  = (f16*)ws;   ws += (size_t)3 * N_EMBED * N_EMBED * 2;        // w_attn^T f16 [2304][768]
  f16*   wpT  = (f16*)ws;   ws += (size_t)N_EMBED * N_EMBED * 2;            // w_proj^T f16 [768][768]
  float* qkv  = (float*)ws; ws += (size_t)BT * 3 * N_EMBED * 4;             // qkv f32
  f16*   Qh   = (f16*)ws;   ws += (size_t)BT * N_EMBED * 2;                 // [B,H,T,64]
  f16*   Kh   = (f16*)ws;   ws += (size_t)BT * N_EMBED * 2;                 // [B,H,T,64]
  f16*   Vt   = (f16*)ws;   ws += (size_t)BT * N_EMBED * 2;                 // [B,H,64,T]
  f16*   Oh   = (f16*)ws;   ws += (size_t)BT * N_EMBED * 2;                 // [B,T,E] f16

  f32_to_f16_kernel<<<(BT * N_EMBED + 255) / 256, 256, 0, stream>>>(x, xh, BT * N_EMBED);
  transpose_f32_to_f16_kernel<<<(N_EMBED * 3 * N_EMBED + 255) / 256, 256, 0, stream>>>(
      w_attn, waT, N_EMBED, 3 * N_EMBED);
  transpose_f32_to_f16_kernel<<<(N_EMBED * N_EMBED + 255) / 256, 256, 0, stream>>>(
      w_proj, wpT, N_EMBED, N_EMBED);

  dim3 gA(3 * N_EMBED / 256, BT / 16);               // 9 x 256 (wave = 16x64 tile)
  gemm_wmma_kernel<<<gA, 128, 0, stream>>>(xh, waT, b_attn, qkv, BT, 3 * N_EMBED, N_EMBED);

  int ropeN = BATCH * SEQ_T * N_HEAD * 32;
  rope_split_kernel<<<(ropeN + 255) / 256, 256, 0, stream>>>(qkv, Qh, Kh, Vt);

  dim3 gAtt(SEQ_T / 16 / 4, BATCH * N_HEAD);          // 32 x 24
  flash_attn_kernel<<<gAtt, 128, 0, stream>>>(Qh, Kh, Vt, Oh);

  dim3 gP(N_EMBED / 256, BT / 16);                    // 3 x 256
  gemm_wmma_kernel<<<gP, 128, 0, stream>>>(Oh, wpT, b_proj, out, BT, N_EMBED, N_EMBED);
}
